// DecoderLayer_50783693308094
// MI455X (gfx1250) — matrix operand order repaired and checked by reading, not compiled, not run on hardware
//
#include <hip/hip_runtime.h>

typedef __bf16  v16bf __attribute__((ext_vector_type(16)));
typedef float   v8f   __attribute__((ext_vector_type(8)));

union AB {
    v16bf v;
    uint4 q[2];
    unsigned u[8];
};

__device__ __forceinline__ unsigned f2bf_u(float x) {
    unsigned u = __float_as_uint(x);
    return (u + 0x7FFFu + ((u >> 16) & 1u)) >> 16;   // round-to-nearest-even
}
__device__ __forceinline__ unsigned pk2(float a, float b) {
    return f2bf_u(a) | (f2bf_u(b) << 16);
}
__device__ __forceinline__ unsigned short lo16(unsigned u) { return (unsigned short)u; }
__device__ __forceinline__ unsigned short hi16(unsigned u) { return (unsigned short)(u >> 16); }

// Async global->LDS 16B copy (gfx1250, tracked by ASYNCcnt). IOFFSET applies
// to both the LDS and global addresses per ISA §10 (op 98 pseudocode).
__device__ __forceinline__ void async_b128(unsigned lds_addr, const void* gaddr) {
    asm volatile("global_load_async_to_lds_b128 %0, %1, off"
                 :: "v"(lds_addr), "v"((unsigned long long)gaddr) : "memory");
}
__device__ __forceinline__ void async_b128_off16(unsigned lds_addr, const void* gaddr) {
    asm volatile("global_load_async_to_lds_b128 %0, %1, off offset:16"
                 :: "v"(lds_addr), "v"((unsigned long long)gaddr) : "memory");
}
__device__ __forceinline__ void wait_async0() {
    asm volatile("s_wait_asynccnt 0x0" ::: "memory");
}
__device__ __forceinline__ unsigned lds_off(const void* p) {
    return (unsigned)(unsigned long long)p;   // generic LDS addr: low 32 bits = LDS offset
}

// ---------------------------------------------------------------------------
// fp32 -> bf16 conversion (8 floats -> uint4 of packed pairs per thread)
// ---------------------------------------------------------------------------
__global__ __launch_bounds__(256)
void cvt_bf16(const float* __restrict__ in, unsigned* __restrict__ out, int n8) {
    int idx = blockIdx.x * 256 + threadIdx.x;
    if (idx >= n8) return;
    const float4* s = (const float4*)in;
    float4 a = s[2 * idx], b = s[2 * idx + 1];
    uint4 o;
    o.x = pk2(a.x, a.y); o.y = pk2(a.z, a.w);
    o.z = pk2(b.x, b.y); o.w = pk2(b.z, b.w);
    ((uint4*)out)[idx] = o;
}

// ---------------------------------------------------------------------------
// LayerNorm: fp32 in -> bf16 (packed pairs) out. One block per row, D=2048.
// ---------------------------------------------------------------------------
__global__ __launch_bounds__(256)
void ln_bf16_kernel(const float* __restrict__ in, const float* __restrict__ g,
                    const float* __restrict__ b, unsigned* __restrict__ out) {
    const int D = 2048;
    const int row = blockIdx.x;
    const float4* src = (const float4*)(in + (size_t)row * D);
    uint2* dst = (uint2*)(out + (size_t)row * (D / 2));
    const float4* g4 = (const float4*)g;
    const float4* b4 = (const float4*)b;

    __shared__ float s1[256], s2[256];
    float4 v0 = src[threadIdx.x];
    float4 v1 = src[threadIdx.x + 256];
    float sum = v0.x + v0.y + v0.z + v0.w + v1.x + v1.y + v1.z + v1.w;
    float sq  = v0.x*v0.x + v0.y*v0.y + v0.z*v0.z + v0.w*v0.w
              + v1.x*v1.x + v1.y*v1.y + v1.z*v1.z + v1.w*v1.w;
    s1[threadIdx.x] = sum; s2[threadIdx.x] = sq;
    __syncthreads();
    for (int off = 128; off > 0; off >>= 1) {
        if (threadIdx.x < off) {
            s1[threadIdx.x] += s1[threadIdx.x + off];
            s2[threadIdx.x] += s2[threadIdx.x + off];
        }
        __syncthreads();
    }
    const float mu = s1[0] * (1.f / D);
    const float var = s2[0] * (1.f / D) - mu * mu;
    const float rs = rsqrtf(var + 1e-5f);

    float4 ga = g4[threadIdx.x],       bb = b4[threadIdx.x];
    float4 gc = g4[threadIdx.x + 256], bd = b4[threadIdx.x + 256];
    uint2 o0, o1;
    o0.x = pk2((v0.x - mu) * rs * ga.x + bb.x, (v0.y - mu) * rs * ga.y + bb.y);
    o0.y = pk2((v0.z - mu) * rs * ga.z + bb.z, (v0.w - mu) * rs * ga.w + bb.w);
    o1.x = pk2((v1.x - mu) * rs * gc.x + bd.x, (v1.y - mu) * rs * gc.y + bd.y);
    o1.y = pk2((v1.z - mu) * rs * gc.z + bd.z, (v1.w - mu) * rs * gc.w + bd.w);
    dst[threadIdx.x] = o0;
    dst[threadIdx.x + 256] = o1;
}

// ---------------------------------------------------------------------------
// bf16 WMMA GEMM: C = A[M,K] @ B[K,N] (+bias) (+res) (relu). A,B bf16.
// FLAGS: bit0 bias, bit1 residual(f32), bit2 relu, bit3 bf16 output.
// 256 threads (8 waves), tile 128x128, K-step 32, double-buffered LDS.
// A tiles staged with async global->LDS DMA; B tiles via register transpose.
// ---------------------------------------------------------------------------
template <int FLAGS>
__global__ __launch_bounds__(256)
void gemm_bf16(const unsigned* __restrict__ A2, const unsigned* __restrict__ B2,
               void* __restrict__ Cv, const float* __restrict__ bias,
               const float* __restrict__ res, int M, int N, int K) {
    __shared__ __align__(16) unsigned As_u[2][128 * 20];  // bf16 pairs, row stride 20 uints
    __shared__ __align__(16) unsigned Bs_u[2][128 * 20];  // transposed [n][k]

    const int tid  = threadIdx.x;
    const int lane = tid & 31;
    const int wave = tid >> 5;
    const int waveM = wave >> 1;          // 0..3
    const int waveN = wave & 1;           // 0..1
    const int m0 = blockIdx.y * 128;
    const int n0 = blockIdx.x * 128;
    const int lr = lane & 15;
    const int kbA = (lane < 16) ? 0 : 4;  // A frag K-offset (uints)
    const int koB = (lane < 16) ? 0 : 4;  // B frag K-offset (uints)
    const int K2 = K >> 1, N2 = N >> 1;

    const int rowA = tid >> 1, halfA = tid & 1;
    const unsigned* aBase = A2 + (size_t)(m0 + rowA) * K2 + halfA * 8;

    auto asyncA = [&](int k0, int buf) {      // 2x16B per thread, DMA to LDS
        unsigned la = lds_off(As_u[buf] + rowA * 20 + halfA * 8);
        const void* ga = (const void*)(aBase + (k0 >> 1));
        async_b128(la, ga);
        async_b128_off16(la, ga);
    };

    uint4 rb[2];
    auto loadB = [&](int k0) {
#pragma unroll
        for (int it = 0; it < 2; ++it) {
            int idx = tid + it * 256;
            int kk = idx >> 4, nseg = idx & 15;
            rb[it] = *(const uint4*)(B2 + (size_t)(k0 + kk) * N2 + (n0 >> 1) + nseg * 4);
        }
    };
    auto storeB = [&](int buf) {
        unsigned short* bs = (unsigned short*)Bs_u[buf];
#pragma unroll
        for (int it = 0; it < 2; ++it) {
            int idx = tid + it * 256;
            int kk = idx >> 4, nseg = idx & 15;
            unsigned short* d = bs + (nseg * 8) * 40 + kk;
            d[0]   = lo16(rb[it].x); d[40]  = hi16(rb[it].x);
            d[80]  = lo16(rb[it].y); d[120] = hi16(rb[it].y);
            d[160] = lo16(rb[it].z); d[200] = hi16(rb[it].z);
            d[240] = lo16(rb[it].w); d[280] = hi16(rb[it].w);
        }
    };

    v8f acc[2][4];
#pragma unroll
    for (int i = 0; i < 2; ++i)
#pragma unroll
        for (int j = 0; j < 4; ++j) {
            v8f z = {0.f,0.f,0.f,0.f,0.f,0.f,0.f,0.f};
            acc[i][j] = z;
        }

    const int nk = K >> 5;
    asyncA(0, 0);
    loadB(0);
    storeB(0);
    wait_async0();                                // A buf0 landed before first barrier
    for (int i = 0; i < nk; ++i) {
        __syncthreads();                          // publishes buffer cur
        const int cur = i & 1;
        if (i + 1 < nk) {
            asyncA((i + 1) << 5, cur ^ 1);        // DMA overlaps WMMA below
            loadB((i + 1) << 5);
        }
        if (i + 2 < nk) {                         // gfx1250 global_prefetch_b8
            __builtin_prefetch(aBase + ((i + 2) << 4), 0, 3);
            __builtin_prefetch(B2 + (size_t)(((i + 2) << 5) + (tid >> 4)) * N2
                                  + (n0 >> 1) + (tid & 15) * 4, 0, 3);
        }

        AB af[2];
#pragma unroll
        for (int mt = 0; mt < 2; ++mt) {
            int row = waveM * 32 + mt * 16 + lr;
            const uint4* p = (const uint4*)(As_u[cur] + row * 20);
            af[mt].q[0] = p[kbA >> 2];
            af[mt].q[1] = p[2 + (kbA >> 2)];
        }
        AB bf[4];
#pragma unroll
        for (int nt = 0; nt < 4; ++nt) {
            int n = waveN * 64 + nt * 16 + lr;
            const uint4* p = (const uint4*)(Bs_u[cur] + n * 20);
            bf[nt].q[0] = p[koB >> 2];
            bf[nt].q[1] = p[(koB >> 2) + 2];
        }
#pragma unroll
        for (int mt = 0; mt < 2; ++mt)
#pragma unroll
            for (int nt = 0; nt < 4; ++nt)
                acc[mt][nt] = __builtin_amdgcn_wmma_f32_16x16x32_bf16(
                    false, af[mt].v, false, bf[nt].v, (short)0, acc[mt][nt], false, false);

        if (i + 1 < nk) {
            storeB(cur ^ 1);
            wait_async0();                        // A(cur^1) in LDS before next barrier
        }
    }

    float* Cf = (float*)Cv;
    unsigned short* Cb = (unsigned short*)Cv;
    const int mh = (lane < 16) ? 0 : 8;
#pragma unroll
    for (int mt = 0; mt < 2; ++mt)
#pragma unroll
        for (int nt = 0; nt < 4; ++nt) {
            int col = n0 + waveN * 64 + nt * 16 + lr;
            float bia = (FLAGS & 1) ? bias[col] : 0.f;
            size_t rbase = (size_t)(m0 + waveM * 32 + mt * 16 + mh);
#pragma unroll
            for (int r = 0; r < 8; ++r) {
                size_t off = (rbase + r) * (size_t)N + col;
                float vl = acc[mt][nt][r] + bia;
                if (FLAGS & 2) vl += res[off];
                if (FLAGS & 4) vl = fmaxf(vl, 0.f);
                if (FLAGS & 8) Cb[off] = (unsigned short)f2bf_u(vl);
                else           Cf[off] = vl;
            }
        }
}

// ---------------------------------------------------------------------------
// Causal attention, bf16 q/k/v in, bf16 out. No-transpose head reshape:
//   head h, index i  <->  matrix row h*128 + i/16, cols (i%16)*128 .. +128
// Block: 128 threads (4 waves), 64 queries; online softmax; WMMA both einsums.
// K tiles staged with async global->LDS DMA (pure byte copy).
// ---------------------------------------------------------------------------
__global__ __launch_bounds__(128)
void attn_kernel(const unsigned* __restrict__ qm2, const unsigned* __restrict__ km2,
                 const unsigned* __restrict__ vm2, unsigned short* __restrict__ om) {
    const int S = 2048, Dm = 2048;
    const float inv = 0.08838834764831845f;   // 1/sqrt(128)

    __shared__ __align__(16) unsigned Ks_u[32 * 68];    // [key][d-pairs], stride 68 uints
    __shared__ __align__(16) unsigned Vs_u[128 * 20];   // transposed [d][key-pairs]
    __shared__ __align__(16) unsigned Ps_u[4 * 16 * 20];
    unsigned short* Vs_s = (unsigned short*)Vs_u;
    unsigned short* Ps_s = (unsigned short*)Ps_u;

    const int tid = threadIdx.x;
    const int lane = tid & 31;
    const int wave = tid >> 5;
    const int lr = lane & 15;
    const int h = blockIdx.y;
    const size_t bb2 = (size_t)blockIdx.z * S * (Dm / 2);   // uints per batch
    const int i0w = blockIdx.x * 64 + wave * 16;            // wave's query base

    // ---- Q fragments: 16 queries == one contiguous row, pure uint4 loads ----
    const unsigned* qrow = qm2 + bb2 + (size_t)(h * 128 + (i0w >> 4)) * (Dm / 2);
    const int kbase = (lane < 16) ? 0 : 8;                  // bf16 elements
    AB qf[4];
#pragma unroll
    for (int t = 0; t < 4; ++t) {
        const uint4* p = (const uint4*)(qrow + lr * 64 + t * 16 + (kbase >> 1));
        qf[t].q[0] = p[0];
        qf[t].q[1] = p[2];     // +8 uints = +16 bf16
    }

    v8f oacc[8];
#pragma unroll
    for (int nt = 0; nt < 8; ++nt) {
        v8f z = {0.f,0.f,0.f,0.f,0.f,0.f,0.f,0.f};
        oacc[nt] = z;
    }
    float m_run[8], l_run[8];
#pragma unroll
    for (int r = 0; r < 8; ++r) { m_run[r] = -3e38f; l_run[r] = 0.f; }

    const int jmax = blockIdx.x * 64 + 64;
    const int ko2 = (lane < 16) ? 0 : 4;

    for (int j0 = 0; j0 < jmax; j0 += 32) {
        __syncthreads();
        {   // stage K tile: async DMA, 4x16B per thread
            const unsigned* src = km2 + bb2 + (size_t)(h * 128 + (j0 >> 4)) * (Dm / 2);
#pragma unroll
            for (int it = 0; it < 4; ++it) {
                int c = tid + it * 128;          // 512 x 16B chunks
                int j = c >> 4, du = (c & 15) * 4;
                async_b128(lds_off(Ks_u + j * 68 + du), (const void*)(src + j * 64 + du));
            }
        }
        {   // stage V tile transposed [d][key] (register path: needs transpose)
            const unsigned* src = vm2 + bb2 + (size_t)(h * 128 + (j0 >> 4)) * (Dm / 2);
#pragma unroll
            for (int it = 0; it < 4; ++it) {
                int c = tid + it * 128;
                int j = c >> 4, dseg = c & 15;   // 8 d-values per chunk
                uint4 v = *(const uint4*)(src + j * 64 + dseg * 4);
                unsigned short* d = Vs_s + (dseg * 8) * 40 + j;
                d[0]   = lo16(v.x); d[40]  = hi16(v.x);
                d[80]  = lo16(v.y); d[120] = hi16(v.y);
                d[160] = lo16(v.z); d[200] = hi16(v.z);
                d[240] = lo16(v.w); d[280] = hi16(v.w);
            }
        }
        wait_async0();                           // K tile landed
        __syncthreads();                         // (compiler adds dscnt wait for V)

        // ---- scores: two 16x16 tiles over 32 keys ----
        v8f st[2];
#pragma unroll
        for (int jt = 0; jt < 2; ++jt) {
            v8f s = {0.f,0.f,0.f,0.f,0.f,0.f,0.f,0.f};
#pragma unroll
            for (int t = 0; t < 4; ++t) {
                AB kf;
                const uint4* p = (const uint4*)(Ks_u + (jt * 16 + lr) * 68);
                int u0 = (t * 16 + ko2) >> 2;     // 16B-aligned
                kf.q[0] = p[u0];
                kf.q[1] = p[u0 + 2];
                s = __builtin_amdgcn_wmma_f32_16x16x32_bf16(
                        false, qf[t].v, false, kf.v, (short)0, s, false, false);
            }
            // mask BEFORE scaling (matches reference), then scale
            int jcol = j0 + jt * 16 + lr;
            int ib = i0w + ((lane < 16) ? 0 : 8);
#pragma unroll
            for (int r = 0; r < 8; ++r) {
                float sv = s[r];
                sv = (jcol > ib + r) ? -1e9f : sv;
                s[r] = sv * inv;
            }
            st[jt] = s;
        }

        // ---- online softmax ----
        float mnew[8], csc[8];
#pragma unroll
        for (int r = 0; r < 8; ++r) {
            float mx = fmaxf(st[0][r], st[1][r]);
#pragma unroll
            for (int d = 1; d < 16; d <<= 1) mx = fmaxf(mx, __shfl_xor(mx, d));
            mnew[r] = fmaxf(m_run[r], mx);
            csc[r] = __expf(m_run[r] - mnew[r]);
        }
#pragma unroll
        for (int jt = 0; jt < 2; ++jt)
#pragma unroll
            for (int r = 0; r < 8; ++r)
                st[jt][r] = __expf(st[jt][r] - mnew[r]);
#pragma unroll
        for (int r = 0; r < 8; ++r) {
            float sm = st[0][r] + st[1][r];
#pragma unroll
            for (int d = 1; d < 16; d <<= 1) sm += __shfl_xor(sm, d);
            l_run[r] = l_run[r] * csc[r] + sm;
            m_run[r] = mnew[r];
        }
#pragma unroll
        for (int nt = 0; nt < 8; ++nt)
#pragma unroll
            for (int r = 0; r < 8; ++r)
                oacc[nt][r] *= csc[r];

        // ---- P (C-layout) -> LDS -> A-fragment layout ----
        {
            unsigned short* pb = Ps_s + wave * 16 * 40;
            int mrow = (lane < 16) ? 0 : 8;
#pragma unroll
            for (int jt = 0; jt < 2; ++jt)
#pragma unroll
                for (int r = 0; r < 8; ++r)
                    pb[(mrow + r) * 40 + jt * 16 + lr] = (unsigned short)f2bf_u(st[jt][r]);
        }
        asm volatile("s_wait_dscnt 0" ::: "memory");   // same-wave LDS RAW, cross-lane
        AB pf;
        {
            const uint4* p = (const uint4*)(Ps_u + wave * 16 * 20 + lr * 20);
            pf.q[0] = p[kbase >> 3];        // uint off 0 or 4 -> uint4 idx 0/1
            pf.q[1] = p[2 + (kbase >> 3)];
        }

        // ---- O += P @ V ----
#pragma unroll
        for (int nt = 0; nt < 8; ++nt) {
            AB vf;
            const uint4* p = (const uint4*)(Vs_u + (nt * 16 + lr) * 20);
            vf.q[0] = p[ko2 >> 2];
            vf.q[1] = p[(ko2 >> 2) + 2];
            oacc[nt] = __builtin_amdgcn_wmma_f32_16x16x32_bf16(
                           false, pf.v, false, vf.v, (short)0, oacc[nt], false, false);
        }
    }

    // ---- normalize + write bf16 (layout b, i, h*128+d) ----
    const int mh = (lane < 16) ? 0 : 8;
    float linv[8];
#pragma unroll
    for (int r = 0; r < 8; ++r) linv[r] = 1.0f / l_run[r];
#pragma unroll
    for (int nt = 0; nt < 8; ++nt) {
        int col = h * 128 + nt * 16 + lr;
#pragma unroll
        for (int r = 0; r < 8; ++r) {
            size_t row = (size_t)blockIdx.z * S + i0w + mh + r;
            om[row * Dm + col] = (unsigned short)f2bf_u(oacc[nt][r] * linv[r]);
        }
    }
}

// ---------------------------------------------------------------------------
extern "C" void kernel_launch(void* const* d_in, const int* in_sizes, int n_in,
                              void* d_out, int out_size, void* d_ws, size_t ws_size,
                              hipStream_t stream) {
    const float* x     = (const float*)d_in[0];
    const float* Wq    = (const float*)d_in[2];
    const float* Wk    = (const float*)d_in[3];
    const float* Wv    = (const float*)d_in[4];
    const float* Wo    = (const float*)d_in[5];
    const float* ln1_g = (const float*)d_in[6];
    const float* ln1_b = (const float*)d_in[7];
    const float* W1    = (const float*)d_in[8];
    const float* b1    = (const float*)d_in[9];
    const float* W2    = (const float*)d_in[10];
    const float* b2    = (const float*)d_in[11];
    const float* ln2_g = (const float*)d_in[12];
    const float* ln2_b = (const float*)d_in[13];

    const int Mrows = 4096;                  // B*S
    char* wsb = (char*)d_ws;
    const size_t MB = 1024 * 1024;
    float*    x1   = (float*)wsb;                        // 32 MB f32 residual stream
    unsigned* h1b  = (unsigned*)(wsb + 32  * MB);        // 16 MB bf16 (reused: attnO)
    unsigned* qb   = (unsigned*)(wsb + 48  * MB);        // 16 MB bf16 (reused: h2)
    unsigned* kb   = (unsigned*)(wsb + 64  * MB);
    unsigned* vb   = (unsigned*)(wsb + 80  * MB);
    unsigned* ff1b = (unsigned*)(wsb + 96  * MB);        // 64 MB bf16 (4096x8192)
    unsigned* wqb  = (unsigned*)(wsb + 160 * MB);        // 8 MB each
    unsigned* wkb  = (unsigned*)(wsb + 168 * MB);
    unsigned* wvb  = (unsigned*)(wsb + 176 * MB);
    unsigned* wob  = (unsigned*)(wsb + 184 * MB);
    unsigned* w1b  = (unsigned*)(wsb + 192 * MB);        // 32 MB each
    unsigned* w2b  = (unsigned*)(wsb + 224 * MB);
    unsigned* attnOb = h1b;
    unsigned* h2b    = qb;
    float*    out  = (float*)d_out;

    // ---- one-time weight conversion to bf16 ----
    cvt_bf16<<<2048, 256, 0, stream>>>(Wq, wqb, 524288);
    cvt_bf16<<<2048, 256, 0, stream>>>(Wk, wkb, 524288);
    cvt_bf16<<<2048, 256, 0, stream>>>(Wv, wvb, 524288);
    cvt_bf16<<<2048, 256, 0, stream>>>(Wo, wob, 524288);
    cvt_bf16<<<8192, 256, 0, stream>>>(W1, w1b, 2097152);
    cvt_bf16<<<8192, 256, 0, stream>>>(W2, w2b, 2097152);

    ln_bf16_kernel<<<Mrows, 256, 0, stream>>>(x, ln1_g, ln1_b, h1b);

    dim3 gP(16, 32);  // N/128=16, M/128=32
    gemm_bf16<8><<<gP, 256, 0, stream>>>(h1b, wqb, qb, nullptr, nullptr, Mrows, 2048, 2048);
    gemm_bf16<8><<<gP, 256, 0, stream>>>(h1b, wkb, kb, nullptr, nullptr, Mrows, 2048, 2048);
    gemm_bf16<8><<<gP, 256, 0, stream>>>(h1b, wvb, vb, nullptr, nullptr, Mrows, 2048, 2048);

    attn_kernel<<<dim3(32, 16, 2), 128, 0, stream>>>(qb, kb, vb, (unsigned short*)attnOb);

    gemm_bf16<2><<<gP, 256, 0, stream>>>(attnOb, wob, x1, nullptr, x, Mrows, 2048, 2048);

    ln_bf16_kernel<<<Mrows, 256, 0, stream>>>(x1, ln2_g, ln2_b, h2b);

    gemm_bf16<13><<<dim3(64, 32), 256, 0, stream>>>(h2b, w1b, ff1b, b1, nullptr, Mrows, 8192, 2048);
    gemm_bf16<3><<<dim3(16, 32), 256, 0, stream>>>(ff1b, w2b, out, b2, x1, Mrows, 2048, 8192);
}
